// LSTM_15436112462586
// MI455X (gfx1250) — matrix operand order repaired and checked by reading, not compiled, not run on hardware
//
#include <hip/hip_runtime.h>

typedef __attribute__((ext_vector_type(16))) _Float16 v16h;
typedef __attribute__((ext_vector_type(8)))  _Float16 v8h;
typedef __attribute__((ext_vector_type(2)))  _Float16 f16x2;
typedef __attribute__((ext_vector_type(8)))  float    v8f;
typedef __attribute__((ext_vector_type(2)))  float    f32x2;

#define T_STEPS 2048
#define BATCH   50
#define FEAT    750
#define HID     100
#define G4      400      // 4*HID
#define KPAD    128      // HID padded to multiple of 32

#define MTILE   128                  // GEMM rows per block (8 WMMA row-tiles)
#define A_ELEMS (MTILE * 32)         // halfs per A buffer
#define B_ELEMS (G4 * 32)            // halfs per B buffer

__device__ __forceinline__ v16h pack16(v8h lo, v8h hi) {
  v16h r;
#pragma unroll
  for (int i = 0; i < 8; ++i) { r[i] = lo[i]; r[i + 8] = hi[i]; }
  return r;
}

// ---------------------------------------------------------------------------
// Phase 1: x_proj[m][n] = sum_k x[m][k] * W_ih[n][k] + (b_ih[n]+b_hh[n])
// M = 102400, N = 400, K = 750 (padded 768). f16 WMMA, f32 accumulate.
// 800 blocks x 800 threads (25 waves). Block: 128 rows x all 400 cols.
// Wave w owns column-tile nt=w; double-buffered LDS, software-pipelined fills.
// ---------------------------------------------------------------------------
__global__ __launch_bounds__(800)
void lstm_xproj_gemm(const float* __restrict__ x,
                     const float* __restrict__ W_ih,
                     const float* __restrict__ b_ih,
                     const float* __restrict__ b_hh,
                     float* __restrict__ xproj) {
  extern __shared__ _Float16 smem[];          // [2][A_ELEMS] + [2][B_ELEMS]
  _Float16* Alds = smem;                      // 16 KB
  _Float16* Blds = smem + 2 * A_ELEMS;        // 51.2 KB (16B-aligned offset)

  const int tid  = threadIdx.x;
  const int wave = tid >> 5;          // 0..24 == nt
  const int lane = tid & 31;
  const int lcol = lane & 15;
  const int hsel = lane >> 4;
  const int m0   = blockIdx.x * MTILE;

  // Staging registers. Units are f32x2 (8B-aligned: row strides 750*4 and k even).
  // A: 128 rows x 16 units = 2048 (3 per thread, guarded). B: 400 x 16 = 6400 (8 exact).
  f32x2 aR[3], bR[8];

  auto loadA = [&](int kb) {
#pragma unroll
    for (int i = 0; i < 3; ++i) {
      int u = tid + i * 800;
      f32x2 v = {0.f, 0.f};
      if (u < 2048) {
        int row = u >> 4, k = kb + (u & 15) * 2;
        if (k < FEAT) v = *(const f32x2*)&x[(size_t)(m0 + row) * FEAT + k];
      }
      aR[i] = v;
    }
  };
  auto loadB = [&](int kb) {
#pragma unroll
    for (int i = 0; i < 8; ++i) {
      int u = tid + i * 800;
      int row = u >> 4, k = kb + (u & 15) * 2;
      f32x2 v = {0.f, 0.f};
      if (k < FEAT) v = *(const f32x2*)&W_ih[(size_t)row * FEAT + k];
      bR[i] = v;
    }
  };
  auto storeAB = [&](int buf) {
#pragma unroll
    for (int i = 0; i < 3; ++i) {
      int u = tid + i * 800;
      if (u < 2048) {
        int row = u >> 4, kk = (u & 15) * 2;
        f16x2 hv; hv[0] = (_Float16)aR[i][0]; hv[1] = (_Float16)aR[i][1];
        *(f16x2*)&Alds[buf * A_ELEMS + row * 32 + kk] = hv;
      }
    }
#pragma unroll
    for (int i = 0; i < 8; ++i) {
      int u = tid + i * 800;
      int row = u >> 4, kk = (u & 15) * 2;
      f16x2 hv; hv[0] = (_Float16)bR[i][0]; hv[1] = (_Float16)bR[i][1];
      *(f16x2*)&Blds[buf * B_ELEMS + row * 32 + kk] = hv;
    }
  };

  v8f acc[8] = {};                    // 8 row-tiles of 16x16 f32
  loadA(0); loadB(0);

  for (int c = 0; c < 24; ++c) {      // 24 K-chunks of 32
    const int buf = c & 1;
    storeAB(buf);
    if (c + 1 < 24) { loadA((c + 1) * 32); loadB((c + 1) * 32); } // hide under WMMA
    __syncthreads();                  // store->read (same buf); read(c)->store(c+2) via barrier(c+1)

    const int nloc = wave * 16 + lcol;
    v8h blo = *(const v8h*)&Blds[buf * B_ELEMS + nloc * 32 + hsel * 16];
    v8h bhi = *(const v8h*)&Blds[buf * B_ELEMS + nloc * 32 + hsel * 16 + 8];
    v16h bfrag = pack16(blo, bhi);
#pragma unroll
    for (int mt = 0; mt < 8; ++mt) {
      const int mloc = mt * 16 + lcol;
      v8h alo = *(const v8h*)&Alds[buf * A_ELEMS + mloc * 32 + hsel * 16];
      v8h ahi = *(const v8h*)&Alds[buf * A_ELEMS + mloc * 32 + hsel * 16 + 8];
      acc[mt] = __builtin_amdgcn_wmma_f32_16x16x32_f16(
          false, pack16(alo, ahi), false, bfrag, (short)0, acc[mt], false, false);
    }
  }

  // Epilogue: bias + store. D layout: M = mt*16 + r + 8*hsel, N = nt*16 + lcol
  const int n = wave * 16 + lcol;
  const float bias = b_ih[n] + b_hh[n];
#pragma unroll
  for (int mt = 0; mt < 8; ++mt) {
#pragma unroll
    for (int r = 0; r < 8; ++r) {
      int m = m0 + mt * 16 + r + 8 * hsel;
      xproj[(size_t)m * G4 + n] = acc[mt][r] + bias;
    }
  }
}

// ---------------------------------------------------------------------------
// Phase 2+3: persistent single-workgroup LSTM scan with fused FC head.
// 1024 threads = 32 waves. Waves 0..24: gates = xproj[t] + h @ W_hh^T via WMMA
// (wave w owns column-tile nt=w; W_hh fragments resident in registers, K 100->128).
// h lives in LDS as f16 in A-fragment layout. All threads: activations, private
// cell state, fused fc dot (shfl reduce + LDS atomics) -> out[t][0..1].
// ---------------------------------------------------------------------------
__global__ __launch_bounds__(1024)
void lstm_scan(const float* __restrict__ xproj,
               const float* __restrict__ W_hh,
               const float* __restrict__ fc_W,
               const float* __restrict__ fc_b,
               float* __restrict__ out) {
  __shared__ __align__(16) _Float16 h16[64 * KPAD];      // 16 KB, padding stays 0
  __shared__ _Float16 gates[BATCH * G4];                 // 40 KB
  __shared__ float outacc[2];

  const int tid  = threadIdx.x;
  const int wave = tid >> 5;
  const int lane = tid & 31;
  const int lcol = lane & 15;
  const int hsel = lane >> 4;

  for (int i = tid; i < 64 * KPAD; i += 1024) h16[i] = (_Float16)0.f;  // h0 = 0
  if (tid == 0) { outacc[0] = 0.f; outacc[1] = 0.f; }

  // Resident W_hh B-fragments: col n = wave*16+lcol, 4 K-chunks of 32 (zero pad >= 100)
  v16h bfr[4] = {};
  if (wave < 25) {
    const int n = wave * 16 + lcol;
    const int run = hsel * 16;
#pragma unroll
    for (int kc = 0; kc < 4; ++kc) {
      v16h f;
#pragma unroll
      for (int i = 0; i < 16; ++i) {
        int k = kc * 32 + run + i;
        f[i] = (k < HID) ? (_Float16)W_hh[n * HID + k] : (_Float16)0.f;
      }
      bfr[kc] = f;
    }
  }

  // Per-thread cell state + fc weights (element e = b*HID + j, up to 5 per thread)
  float cpriv[5], fw0[5], fw1[5];
#pragma unroll
  for (int s = 0; s < 5; ++s) {
    cpriv[s] = 0.f;
    int e = tid + s * 1024;
    if (e < BATCH * HID) { fw0[s] = fc_W[e]; fw1[s] = fc_W[BATCH * HID + e]; }
    else                 { fw0[s] = 0.f;     fw1[s] = 0.f; }
  }
  const float fb0 = fc_b[0], fb1 = fc_b[1];

  __syncthreads();

  for (int t = 0; t < T_STEPS; ++t) {
    // Prefetch next step's x_proj block (50*400*4B = 625 cachelines of 128B, 640 issued)
    if (t + 1 < T_STEPS) {
      int line = wave * 32 + lane;
      if (line < 640) {
        const char* p = (const char*)xproj +
                        (size_t)(t + 1) * BATCH * G4 * sizeof(float) +
                        (size_t)line * 128;
        __builtin_prefetch(p, 0, 1);
      }
    }

    if (wave < 25) {
      const int n = wave * 16 + lcol;
      v8f acc[4];
#pragma unroll
      for (int mt = 0; mt < 4; ++mt) {                 // init C with xproj[t]
#pragma unroll
        for (int r = 0; r < 8; ++r) {
          // b = mt*16 + r + 8*hsel; b < 50 always for mt<3; for mt==3: hsel==0 && r<2
          if (mt < 3 || (hsel == 0 && r < 2)) {
            int b = mt * 16 + r + 8 * hsel;
            acc[mt][r] = xproj[((size_t)t * BATCH + b) * G4 + n];
          } else {
            acc[mt][r] = 0.f;
          }
        }
      }
#pragma unroll
      for (int kc = 0; kc < 4; ++kc) {
#pragma unroll
        for (int mt = 0; mt < 4; ++mt) {
          const int mrow = mt * 16 + lcol;
          const int r1 = kc * 32 + hsel * 16;
          v8h alo = *(const v8h*)&h16[mrow * KPAD + r1];
          v8h ahi = *(const v8h*)&h16[mrow * KPAD + r1 + 8];
          acc[mt] = __builtin_amdgcn_wmma_f32_16x16x32_f16(
              false, pack16(alo, ahi), false, bfr[kc], (short)0, acc[mt], false, false);
        }
      }
#pragma unroll
      for (int mt = 0; mt < 4; ++mt) {                 // publish gates
#pragma unroll
        for (int r = 0; r < 8; ++r) {
          if (mt < 3 || (hsel == 0 && r < 2)) {
            int b = mt * 16 + r + 8 * hsel;
            gates[b * G4 + n] = (_Float16)acc[mt][r];
          }
        }
      }
    }
    __syncthreads();

    // Elementwise LSTM cell + fused fc partial
    float p0 = 0.f, p1 = 0.f;
#pragma unroll
    for (int s = 0; s < 5; ++s) {
      int e = tid + s * 1024;
      if (e < BATCH * HID) {
        int b = e / HID, j = e % HID;
        float gi = (float)gates[b * G4 + j];
        float gf = (float)gates[b * G4 + HID + j];
        float gg = (float)gates[b * G4 + 2 * HID + j];
        float go = (float)gates[b * G4 + 3 * HID + j];
        float i_ = 1.f / (1.f + __expf(-gi));
        float f_ = 1.f / (1.f + __expf(-gf));
        float g_ = tanhf(gg);
        float o_ = 1.f / (1.f + __expf(-go));
        float c  = f_ * cpriv[s] + i_ * g_;
        cpriv[s] = c;
        float h  = o_ * tanhf(c);
        h16[b * KPAD + j] = (_Float16)h;               // feeds next step's A
        p0 += h * fw0[s];
        p1 += h * fw1[s];
      }
    }
    // wave32 tree-reduce, then one LDS atomic per wave
#pragma unroll
    for (int off = 16; off > 0; off >>= 1) {
      p0 += __shfl_xor(p0, off, 32);
      p1 += __shfl_xor(p1, off, 32);
    }
    if (lane == 0) { atomicAdd(&outacc[0], p0); atomicAdd(&outacc[1], p1); }
    __syncthreads();
    if (tid == 0) {
      out[t * 2 + 0] = outacc[0] + fb0;
      out[t * 2 + 1] = outacc[1] + fb1;
      outacc[0] = 0.f; outacc[1] = 0.f;
    }
    __syncthreads();
  }
}

// ---------------------------------------------------------------------------
extern "C" void kernel_launch(void* const* d_in, const int* in_sizes, int n_in,
                              void* d_out, int out_size, void* d_ws, size_t ws_size,
                              hipStream_t stream) {
  const float* x    = (const float*)d_in[0];
  const float* W_ih = (const float*)d_in[1];
  const float* W_hh = (const float*)d_in[2];
  const float* b_ih = (const float*)d_in[3];
  const float* b_hh = (const float*)d_in[4];
  const float* fc_W = (const float*)d_in[5];
  const float* fc_b = (const float*)d_in[6];
  float* out   = (float*)d_out;
  float* xproj = (float*)d_ws;   // 102400 * 400 floats = 163.84 MB scratch

  const size_t shmem = (size_t)(2 * A_ELEMS + 2 * B_ELEMS) * sizeof(_Float16); // 67.6 KB
  lstm_xproj_gemm<<<102400 / MTILE, 800, shmem, stream>>>(x, W_ih, b_ih, b_hh, xproj);
  lstm_scan<<<1, 1024, 0, stream>>>(xproj, W_hh, fc_W, fc_b, out);
}
